// _FlashRNNTritonFusedLayer_89790586290287
// MI455X (gfx1250) — matrix operand order repaired and checked by reading, not compiled, not run on hardware
//
#include <hip/hip_runtime.h>
#include <hip/hip_bf16.h>
#include <stdint.h>

// Problem constants (fixed by the reference).
#define B_   16
#define T_   1024
#define NG_  4
#define D_   768
#define DW   16            // d-columns owned per workgroup
#define NWG  (D_ / DW)     // 48 persistent workgroups
#define KB_  (D_ / 32)     // 24 k-blocks of 32 (bf16 WMMA K)
#define KBH  (KB_ / 2)     // 12 k-blocks per K-half (per wave)

typedef __attribute__((ext_vector_type(16))) __bf16 v16bf;
typedef __attribute__((ext_vector_type(8)))  __bf16 v8bf;
typedef __attribute__((ext_vector_type(8)))  float  v8f;
typedef __attribute__((ext_vector_type(4)))  unsigned int v4u;
typedef __attribute__((ext_vector_type(8)))  int v8i;
typedef __attribute__((ext_vector_type(4)))  int v4i;

// Tensor Data Mover for the per-step h broadcast (6-arg builtin on this
// toolchain). Fallback: cooperative uint4 copy.
#if defined(__HIP_DEVICE_COMPILE__) && defined(__gfx1250__) && __has_builtin(__builtin_amdgcn_tensor_load_to_lds)
#define USE_TDM 1
#else
#define USE_TDM 0
#endif

__device__ __forceinline__ float sigf(float x) {
  return 1.0f / (1.0f + __expf(-x));
}

// Sense-reversing grid barrier. Monotonic targets within one launch; the
// counter/sense words are zeroed by hipMemsetAsync before every launch so
// graph replays see identical state.
__device__ __forceinline__ void grid_barrier(unsigned* cnt, unsigned* sense,
                                             unsigned target) {
  __syncthreads();
  if (threadIdx.x == 0) {
    __threadfence();
    unsigned prev = atomicAdd(cnt, 1u);
    if (prev == (unsigned)(NWG - 1)) {
      atomicExch(cnt, 0u);
      __threadfence();
      atomicExch(sense, target);
    } else {
      while (__hip_atomic_load(sense, __ATOMIC_ACQUIRE,
                               __HIP_MEMORY_SCOPE_AGENT) < target) {
#if defined(__HIP_DEVICE_COMPILE__)
        __builtin_amdgcn_s_sleep(1);
#endif
      }
    }
  }
  __syncthreads();
}

__global__ __launch_bounds__(256)
void flashrnn_persistent(const float* __restrict__ states,   // [2,B,1,D]
                         const float* __restrict__ Wx,       // [B,T,4,1,D]
                         const float* __restrict__ R,        // [4,1,D,D]
                         const float* __restrict__ bias,     // [4,1,D]
                         float* __restrict__ out,            // [2,B,T+1,1,D]
                         uint16_t* __restrict__ hbuf0_,
                         uint16_t* __restrict__ hbuf1_,
                         unsigned* __restrict__ bar_cnt,
                         unsigned* __restrict__ bar_sense) {
  const int tid  = threadIdx.x;
  const int lane = tid & 31;          // wave32
  const int wave = tid >> 5;          // 8 waves
  const int g    = wave & 3;          // gate owned by this wave
  const int kh   = wave >> 2;         // K-half owned by this wave
  const int d0   = blockIdx.x * DW;   // this WG's d-slice

  __shared__ __align__(16) __bf16 hstage[B_ * D_];   // 24 KB staged h
  __shared__ float gbuf[NG_][B_ * DW];               // 4 KB gate tiles / partials

  __bf16* hb0 = (__bf16*)hbuf0_;
  __bf16* hb1 = (__bf16*)hbuf1_;

  // Per-thread elementwise identity: thread <-> (batch, local d).
  const int eb = tid >> 4;            // 0..15 batch
  const int ed = tid & 15;            // 0..15 local d
  const int dg = d0 + ed;             // global d

  // ---- Init: c0 register, t=0 outputs, h0 -> ping-pong buffer 0, bias regs.
  float c_reg = states[(B_ + eb) * D_ + dg];
  {
    float h0v = states[eb * D_ + dg];
    out[((size_t)eb * (T_ + 1) + 0) * D_ + dg]        = h0v;
    out[((size_t)(B_ + eb) * (T_ + 1) + 0) * D_ + dg] = c_reg;
    hb0[eb * D_ + dg] = (__bf16)h0v;
  }
  float bias_r[NG_];
#pragma unroll
  for (int gg = 0; gg < NG_; ++gg) bias_r[gg] = bias[gg * D_ + dg];

  // ---- Load persistent B-fragments of R (bf16) into registers: 12 x v16bf.
  // B fragment (32xK x 16xN), contiguous k order used here: lane l holds N = l%16; element e <-> K =
  // kb*32 + e + 16*(l/16).
  const int h2   = lane >> 4;
  const int nloc = lane & 15;
  v16bf bfrag[KBH];
#pragma unroll
  for (int j = 0; j < KBH; ++j) {
    const int kb = kh * KBH + j;
    const int K0 = kb * 32 + 16 * h2;
    const int n  = d0 + nloc;
    v16bf f;
#pragma unroll
    for (int e = 0; e < 16; ++e)
      f[e] = (__bf16)R[((size_t)g * D_ + (K0 + e)) * D_ + n];
    bfrag[j] = f;
  }

  grid_barrier(bar_cnt, bar_sense, 1u);   // hbuf0 fully populated everywhere

  const int mrow = lane & 15;             // A-matrix row (batch) per lane
  for (int t = 0; t < T_; ++t) {
    const __bf16* hsrc = (t & 1) ? hb1 : hb0;
    __bf16*       hdst = (t & 1) ? hb0 : hb1;

    // ---- Stage h (16x768 bf16, contiguous 24 KB) into LDS.
#if USE_TDM
    if (wave == 0) {
      const uint64_t ga   = (uint64_t)(uintptr_t)hsrc;
      const uint32_t lds  = (uint32_t)(uintptr_t)(&hstage[0]);  // LDS aperture low bits
      v4u g0;
      g0.x = 1u;                                        // count=1, user desc
      g0.y = lds;                                       // lds_addr
      g0.z = (uint32_t)(ga & 0xFFFFFFFFu);              // global_addr[31:0]
      g0.w = (uint32_t)((ga >> 32) & 0x01FFFFFFu) | (2u << 30);  // [56:32] | type=2
      v8i g1;
      g1[0] = 0x00010000;            // wg_mask=0, data_size=1 (2 bytes)
      g1[1] = (int)(0x3000u << 16);  // tensor_dim0 = 12288 (bits 79:48 lo half)
      g1[2] = 0x00010000;            // tensor_dim0 hi = 0, tensor_dim1 = 1
      g1[3] = (int)(0x3000u << 16);  // tile_dim0 = 12288
      g1[4] = 1;                     // tile_dim1 = 1, tile_dim2 = 0
      g1[5] = 12288;                 // tensor_dim0_stride lo
      g1[6] = 0;
      g1[7] = 0;
      v4i gz4 = {0, 0, 0, 0};
      v8i gz8 = {0, 0, 0, 0, 0, 0, 0, 0};
      __builtin_amdgcn_tensor_load_to_lds(g0, g1, gz4, gz4, gz8, 0);
      __builtin_amdgcn_s_wait_tensorcnt(0);
    }
#else
    {
      const uint4* s4 = (const uint4*)hsrc;
      uint4*       t4 = (uint4*)hstage;
#pragma unroll
      for (int i = 0; i < (B_ * D_ * 2 / 16) / 256; ++i)   // 6 x uint4 per thread
        t4[tid + i * 256] = s4[tid + i * 256];
    }
#endif
    __syncthreads();

    // ---- 12-deep WMMA accumulation over this wave's K-half.
    // A bf16 16x32 (contiguous k, as B): lane l row M=l%16; elems 0..15 <-> K=kb*32+16*(l/16)+e:
    // two contiguous 16B LDS reads.
    // Software-pipelined: A fragment for block j+1 is loaded into the alternate
    // buffer before the WMMA for block j issues, so the ds_load latency hides
    // under the WMMA chain instead of serializing with it.
    const __bf16* abase = &hstage[mrow * D_ + 16 * h2];
    v8bf alo[2], ahi[2];
    {
      const int kb0 = kh * KBH;
      alo[0] = *(const v8bf*)&abase[kb0 * 32];
      ahi[0] = *(const v8bf*)&abase[kb0 * 32 + 8];
    }
    v8f acc = {};
#pragma unroll
    for (int j = 0; j < KBH; ++j) {
      const int cur = j & 1;
      if (j + 1 < KBH) {
        const int kbn = kh * KBH + j + 1;
        alo[cur ^ 1] = *(const v8bf*)&abase[kbn * 32];
        ahi[cur ^ 1] = *(const v8bf*)&abase[kbn * 32 + 8];
      }
      v16bf a;
#pragma unroll
      for (int e = 0; e < 8; ++e) { a[e] = alo[cur][e]; a[8 + e] = ahi[cur][e]; }
      acc = __builtin_amdgcn_wmma_f32_16x16x32_bf16(
          false, a, false, bfrag[j], (short)0, acc, false, false);
    }

    // ---- Reduce the two K-halves through LDS (C layout: VGPR r, lane l ->
    // M = r + 8*(l/16), N = l%16).
    const int mb = h2 * 8;
    if (kh == 1) {
#pragma unroll
      for (int r = 0; r < 8; ++r) gbuf[g][(mb + r) * DW + nloc] = acc[r];
    }
    __syncthreads();
    if (kh == 0) {
#pragma unroll
      for (int r = 0; r < 8; ++r) gbuf[g][(mb + r) * DW + nloc] += acc[r];
    }
    __syncthreads();

    // ---- Elementwise cell update: one thread per (b, d).
    const float* wxp = &Wx[((size_t)(eb * T_ + t) * NG_) * D_ + dg];
    const float pi = gbuf[0][eb * DW + ed] + wxp[0 * D_] + bias_r[0];
    const float pf = gbuf[1][eb * DW + ed] + wxp[1 * D_] + bias_r[1];
    const float pz = gbuf[2][eb * DW + ed] + wxp[2 * D_] + bias_r[2];
    const float po = gbuf[3][eb * DW + ed] + wxp[3 * D_] + bias_r[3];
    const float ig = sigf(pi), fg = sigf(pf), og = sigf(po);
    const float zg = tanhf(pz);
    c_reg = fg * c_reg + ig * zg;
    const float h_new = og * tanhf(c_reg);

    out[((size_t)eb * (T_ + 1) + (t + 1)) * D_ + dg]        = h_new;
    out[((size_t)(B_ + eb) * (T_ + 1) + (t + 1)) * D_ + dg] = c_reg;
    hdst[eb * D_ + dg] = (__bf16)h_new;

    // Prefetch next step's Wx gates while we sit in the barrier.
    if (t + 1 < T_) {
      const float* nx = &Wx[((size_t)(eb * T_ + (t + 1)) * NG_) * D_ + dg];
      __builtin_prefetch(nx + 0 * D_, 0, 1);
      __builtin_prefetch(nx + 1 * D_, 0, 1);
      __builtin_prefetch(nx + 2 * D_, 0, 1);
      __builtin_prefetch(nx + 3 * D_, 0, 1);
    }

    __threadfence();
    grid_barrier(bar_cnt, bar_sense, (unsigned)(t + 2));
  }
}

extern "C" void kernel_launch(void* const* d_in, const int* in_sizes, int n_in,
                              void* d_out, int out_size, void* d_ws, size_t ws_size,
                              hipStream_t stream) {
  (void)in_sizes; (void)n_in; (void)out_size; (void)ws_size;
  const float* states = (const float*)d_in[0];
  const float* Wx     = (const float*)d_in[1];
  const float* R      = (const float*)d_in[2];
  const float* bias   = (const float*)d_in[3];
  float* out = (float*)d_out;

  // d_ws layout: [0,256) barrier words; then two 24 KB bf16 h ping-pong buffers.
  // Requires ws_size >= 256 + 2*16*768*2 = 49408 bytes.
  unsigned char* ws = (unsigned char*)d_ws;
  unsigned* bar  = (unsigned*)ws;
  uint16_t* hb0  = (uint16_t*)(ws + 256);
  uint16_t* hb1  = (uint16_t*)(ws + 256 + B_ * D_ * sizeof(uint16_t));

  (void)hipMemsetAsync(d_ws, 0, 256, stream);  // deterministic barrier state per launch
  flashrnn_persistent<<<dim3(NWG), dim3(256), 0, stream>>>(
      states, Wx, R, bias, out, hb0, hb1, bar, bar + 1);
}
